// MaskedCharacterPredictor_39058432590237
// MI455X (gfx1250) — matrix operand order repaired and checked by reading, not compiled, not run on hardware
//
#include <hip/hip_runtime.h>
#include <hip/hip_bf16.h>
#include <stdint.h>

// ---------------- problem constants ----------------
#define B_    512
#define L_    45
#define BL_   (B_*L_)        // 23040
#define E_    128
#define H_    1024
#define G4H_  (4*H_)         // 4096
#define CTX_  32
#define CNN_  64
#define PR_   26
#define DIN_  160            // E_ + CTX_ (multiple of 32)
#define KCAT_ (DIN_+H_)      // 1184 (multiple of 32)
#define DECK_ 2080           // 2048 encoded + 26 prior, padded to mult of 32

typedef _Float16 half_t;
typedef __attribute__((ext_vector_type(16))) _Float16 v16h;
typedef __attribute__((ext_vector_type(8)))  _Float16 v8h;
typedef __attribute__((ext_vector_type(8)))  float    v8f;
typedef int v4i_ __attribute__((vector_size(16)));

typedef __attribute__((address_space(1))) v4i_* gbl_v4i_p;
typedef __attribute__((address_space(3))) v4i_* lds_v4i_p;

union HV16 { v16h v; v8h h[2]; };

// CDNA5 async global->LDS path (ASYNCcnt); fall back to sync staging if the
// toolchain doesn't declare the gfx1250 builtins.
#if defined(__has_builtin)
#if __has_builtin(__builtin_amdgcn_global_load_async_to_lds_b128) && \
    __has_builtin(__builtin_amdgcn_s_wait_asynccnt)
#define USE_ASYNC_LDS 1
#endif
#endif
#ifndef USE_ASYNC_LDS
#define USE_ASYNC_LDS 0
#endif

__device__ __forceinline__ float sigmoidf_(float x){ return 1.f/(1.f+__expf(-x)); }

// Stage a [BN cols x 32 K] f16 W tile into LDS. 128 threads/block.
// LDS layout: Bs[col*32 + k], so a B fragment (16 contiguous K for one col)
// is one 32B LDS read. Wrow0 = W + nBase*K + k0 (W is [N,K] row-major).
template<int BN>
__device__ __forceinline__ void stageB(const half_t* __restrict__ Wrow0, int K,
                                       half_t* Bs){
    constexpr int CH = BN / 32;              // 16B chunks per thread
    #pragma unroll
    for (int i = 0; i < CH; ++i){
        int chunk = threadIdx.x + i*128;
        int col  = chunk >> 2;
        int koff = (chunk & 3) * 8;
        const half_t* src = Wrow0 + (size_t)col*K + koff;
        half_t* dst = Bs + col*32 + koff;
#if USE_ASYNC_LDS
        void* s_ = (void*)src;               // drop const
        void* d_ = (void*)dst;
        __builtin_amdgcn_global_load_async_to_lds_b128(
            (gbl_v4i_p)s_, (lds_v4i_p)d_, 0, 0);
#else
        *(v8h*)dst = *(const v8h*)src;
#endif
    }
#if USE_ASYNC_LDS
    __builtin_amdgcn_s_wait_asynccnt(0);
#endif
}

// ---------------- utility kernels ----------------
__global__ void k_fill0(uint32_t* p, int nWords){
    int i = blockIdx.x*blockDim.x + threadIdx.x;
    if (i < nWords) p[i] = 0u;
}

// f32 -> f16 copy with dst pitch / column offset (used to repack weights)
__global__ void k_cvt(const float* __restrict__ src, half_t* __restrict__ dst,
                      int rows, int srcK, int dstLd, int dstOff){
    int i = blockIdx.x*blockDim.x + threadIdx.x;
    int tot = rows*srcK;
    if (i >= tot) return;
    int r = i / srcK, c = i - r*srcK;
    dst[(size_t)r*dstLd + dstOff + c] = (half_t)src[i];
}

// enc_in = concat(char_emb[word_state], ctx_emb[position_context]) as f16 [BL,160]
__global__ void k_embed(const int* __restrict__ ws, const int* __restrict__ pc,
                        const float* __restrict__ cemb, const float* __restrict__ xemb,
                        half_t* __restrict__ encIn){
    int i = blockIdx.x*blockDim.x + threadIdx.x;
    if (i >= BL_*DIN_) return;
    int row = i / DIN_, c = i - row*DIN_;
    float v = (c < E_) ? cemb[ws[row]*E_ + c] : xemb[pc[row]*CTX_ + (c - E_)];
    encIn[(size_t)row*DIN_ + c] = (half_t)v;
}

// conv1: y[b,o,l] = relu(b + sum_{i,t} emb[b,i,l+t-1]*w[o,i,t]); emb via lookup
__global__ void k_conv1(const int* __restrict__ ws, const float* __restrict__ cemb,
                        const float* __restrict__ w, const float* __restrict__ bias,
                        float* __restrict__ c1){
    int i = blockIdx.x*blockDim.x + threadIdx.x;
    if (i >= BL_*CNN_) return;
    int row = i / CNN_, o = i - row*CNN_;
    int bb = row / L_, l = row - bb*L_;
    float acc = bias[o];
    #pragma unroll
    for (int t = 0; t < 3; ++t){
        int lp = l + t - 1;
        if (lp < 0 || lp >= L_) continue;
        const float* e  = cemb + (size_t)ws[bb*L_ + lp]*E_;
        const float* wp = w + (size_t)o*E_*3 + t;
        for (int k = 0; k < E_; ++k) acc += e[k] * wp[k*3];
    }
    c1[i] = fmaxf(acc, 0.f);
}

__global__ void k_conv2(const float* __restrict__ x, const float* __restrict__ w,
                        const float* __restrict__ bias, float* __restrict__ y){
    int i = blockIdx.x*blockDim.x + threadIdx.x;
    if (i >= BL_*CNN_) return;
    int row = i / CNN_, o = i - row*CNN_;
    int bb = row / L_, l = row - bb*L_;
    float acc = bias[o];
    #pragma unroll
    for (int t = 0; t < 3; ++t){
        int lp = l + t - 1;
        if (lp < 0 || lp >= L_) continue;
        const float* xp = x + (size_t)(bb*L_ + lp)*CNN_;
        const float* wp = w + (size_t)o*CNN_*3 + t;
        for (int k = 0; k < CNN_; ++k) acc += xp[k] * wp[k*3];
    }
    y[i] = fmaxf(acc, 0.f);
}

// prior MLP: [is_blank, blank_count/L, cnn(64)] -> 32 relu -> 26; write f16 into
// dec_in columns 2048..2073, zero-pad 2074..2079
__global__ void k_prior(const int* __restrict__ ws, const int* __restrict__ bc,
                        const float* __restrict__ cnn,
                        const float* __restrict__ w1, const float* __restrict__ b1,
                        const float* __restrict__ w2, const float* __restrict__ b2,
                        half_t* __restrict__ decIn){
    int row = blockIdx.x*blockDim.x + threadIdx.x;
    if (row >= BL_) return;
    int bb = row / L_;
    float x0 = (ws[row] == 0) ? 1.f : 0.f;
    float x1 = (float)bc[bb] / (float)L_;
    const float* cf = cnn + (size_t)row*CNN_;
    float h[32];
    #pragma unroll
    for (int j = 0; j < 32; ++j){
        const float* wj = w1 + j*66;
        float a = b1[j] + wj[0]*x0 + wj[1]*x1;
        for (int k = 0; k < CNN_; ++k) a += wj[2+k]*cf[k];
        h[j] = fmaxf(a, 0.f);
    }
    half_t* d = decIn + (size_t)row*DECK_ + 2048;
    #pragma unroll
    for (int n = 0; n < PR_; ++n){
        float a = b2[n];
        const float* wn = w2 + n*32;
        #pragma unroll
        for (int j = 0; j < 32; ++j) a += wn[j]*h[j];
        d[n] = (half_t)a;
    }
    #pragma unroll
    for (int n = PR_; n < 32; ++n) d[n] = (half_t)0.f;
}

// routing compaction: append row ids per position type (1,2,3)
__global__ void k_route(const int* __restrict__ pc, int* __restrict__ cnt,
                        int* __restrict__ lists){
    int row = blockIdx.x*blockDim.x + threadIdx.x;
    if (row >= BL_) return;
    int t = pc[row];
    if (t >= 1 && t <= 3){
        int i = atomicAdd(&cnt[t], 1);
        lists[(size_t)(t-1)*BL_ + i] = row;
    }
}

// ---------------- LSTM gates GEMM (WMMA) ----------------
// gates[512,4096] = [x_t | h_{t-1}] (f16, K=1184) x Wcat^T (f16 [4096,1184]) + bias
// block = 128 threads = 4 waves; block tile 64(M) x BN(N); wave tile 16 x BN
template<int BN>
__global__ void k_lstm_gates(const half_t* __restrict__ encIn,
                             const half_t* __restrict__ hprev,
                             const half_t* __restrict__ Wcat,
                             const float* __restrict__ bias,
                             float* __restrict__ gates, int tstep){
    constexpr int NT = BN / 16;
    __shared__ __align__(32) half_t Bs[BN*32];
    const int lane = threadIdx.x & 31;
    const int wave = threadIdx.x >> 5;
    const int laneLo = lane & 15;
    const int grp = lane >> 4;                 // 0 or 1
    const int nBase = blockIdx.x * BN;
    const int mBase = blockIdx.y * 64 + wave * 16;
    const int bRow = mBase + laneLo;           // batch row, < 512

    v8f zero8 = {0.f,0.f,0.f,0.f,0.f,0.f,0.f,0.f};
    v8f acc[NT];
    #pragma unroll
    for (int tn = 0; tn < NT; ++tn) acc[tn] = zero8;

    for (int k0 = 0; k0 < KCAT_; k0 += 32){
        __syncthreads();
        stageB<BN>(Wcat + (size_t)nBase*KCAT_ + k0, KCAT_, Bs);
        __syncthreads();
        // A fragment: lane<16 holds K {0..15}, lane>=16 holds K {16..31}
        const half_t* aPtr;
        if (k0 < DIN_) aPtr = encIn + (size_t)(bRow*L_ + tstep)*DIN_ + k0;
        else           aPtr = hprev + (size_t)bRow*H_ + (k0 - DIN_);
        __builtin_prefetch(aPtr + 32, 0, 1);
        HV16 a;
        a.h[0] = *(const v8h*)(aPtr + 16*grp);
        a.h[1] = *(const v8h*)(aPtr + 16*grp + 8);
        #pragma unroll
        for (int tn = 0; tn < NT; ++tn){
            int colL = tn*16 + laneLo;
            v16h bf = *(const v16h*)&Bs[colL*32 + grp*16];
            acc[tn] = __builtin_amdgcn_wmma_f32_16x16x32_f16(
                false, a.v, false, bf, (short)0, acc[tn], false, false);
        }
    }
    // C layout: lane col = laneLo; vgpr r -> row mBase + r + grp*8
    #pragma unroll
    for (int tn = 0; tn < NT; ++tn){
        int col = nBase + tn*16 + laneLo;
        float bv = bias[col];
        #pragma unroll
        for (int r = 0; r < 8; ++r){
            int row = mBase + r + grp*8;
            gates[(size_t)row*G4H_ + col] = acc[tn][r] + bv;
        }
    }
}

// pointwise LSTM cell; writes h (f16) to ping-pong buffer and into dec_in
__global__ void k_lstm_point(const float* __restrict__ gates,
                             float* __restrict__ cst,
                             half_t* __restrict__ hout,
                             half_t* __restrict__ decIn,
                             int tstep, int dirOff){
    int i = blockIdx.x*blockDim.x + threadIdx.x;
    if (i >= B_*H_) return;
    int bb = i >> 10, n = i & (H_-1);
    const float* g = gates + (size_t)bb*G4H_;
    float ig = g[n], fg = g[H_+n], gg = g[2*H_+n], og = g[3*H_+n];
    float c = sigmoidf_(fg)*cst[i] + sigmoidf_(ig)*tanhf(gg);
    float h = sigmoidf_(og)*tanhf(c);
    cst[i] = c;
    hout[i] = (half_t)h;
    decIn[(size_t)(bb*L_ + tstep)*DECK_ + dirOff + n] = (half_t)h;
}

// ---------------- decoder GEMM (WMMA, routed) ----------------
// C = act(A x W^T + bias). A rows optionally gathered via list; final layer
// scatters f32 by list with N=26 column mask. M read from device counter.
template<int BN, bool GATHER, bool RELU, bool FINAL>
__global__ void k_dec_gemm(const half_t* __restrict__ A, int lda,
                           const half_t* __restrict__ W, int K,
                           const float* __restrict__ bias,
                           half_t* __restrict__ Cf16, int ldc,
                           float* __restrict__ outF32, int N,
                           const int* __restrict__ list,
                           const int* __restrict__ cntPtr){
    const int M = *cntPtr;
    if ((int)blockIdx.y * 64 >= M) return;     // whole-block exit (barrier-safe)

    constexpr int NT = BN / 16;
    __shared__ __align__(32) half_t Bs[BN*32];
    const int lane = threadIdx.x & 31;
    const int wave = threadIdx.x >> 5;
    const int laneLo = lane & 15;
    const int grp = lane >> 4;
    const int nBase = blockIdx.x * BN;
    const int mBase = blockIdx.y * 64 + wave * 16;
    const int mRow = mBase + laneLo;

    int srcRow;
    if (GATHER) srcRow = (mRow < M) ? list[mRow] : list[0];
    else        srcRow = (mRow < M) ? mRow : 0;
    const half_t* aRow = A + (size_t)srcRow * lda;

    v8f zero8 = {0.f,0.f,0.f,0.f,0.f,0.f,0.f,0.f};
    v8f acc[NT];
    #pragma unroll
    for (int tn = 0; tn < NT; ++tn) acc[tn] = zero8;

    for (int k0 = 0; k0 < K; k0 += 32){
        __syncthreads();
        stageB<BN>(W + (size_t)nBase*K + k0, K, Bs);
        __syncthreads();
        __builtin_prefetch(aRow + k0 + 32, 0, 1);
        HV16 a;
        a.h[0] = *(const v8h*)(aRow + k0 + 16*grp);
        a.h[1] = *(const v8h*)(aRow + k0 + 16*grp + 8);
        #pragma unroll
        for (int tn = 0; tn < NT; ++tn){
            int colL = tn*16 + laneLo;
            v16h bf = *(const v16h*)&Bs[colL*32 + grp*16];
            acc[tn] = __builtin_amdgcn_wmma_f32_16x16x32_f16(
                false, a.v, false, bf, (short)0, acc[tn], false, false);
        }
    }
    #pragma unroll
    for (int tn = 0; tn < NT; ++tn){
        int col = nBase + tn*16 + laneLo;
        float bv = (col < N) ? bias[col] : 0.f;
        #pragma unroll
        for (int r = 0; r < 8; ++r){
            int row = mBase + r + grp*8;
            float v = acc[tn][r] + bv;
            if (RELU) v = fmaxf(v, 0.f);
            if (FINAL){
                if (row < M && col < N)
                    outF32[(size_t)list[row]*N + col] = v;
            } else {
                if (row < M)
                    Cf16[(size_t)row*ldc + col] = (half_t)v;
            }
        }
    }
}

// ---------------- host orchestration ----------------
static inline int cdiv_(int a, int b){ return (a + b - 1) / b; }

extern "C" void kernel_launch(void* const* d_in, const int* in_sizes, int n_in,
                              void* d_out, int out_size, void* d_ws, size_t ws_size,
                              hipStream_t stream) {
    (void)in_sizes; (void)n_in; (void)out_size; (void)ws_size;

    // ---- inputs (dict order, tuples flattened) ----
    const int*   ws_i  = (const int*)  d_in[0];
    const int*   pc_i  = (const int*)  d_in[1];
    // d_in[2] word_length (unused), d_in[4] next_to_vowel (unused)
    const int*   bc_i  = (const int*)  d_in[3];
    const float* cembF = (const float*)d_in[5];
    const float* xembF = (const float*)d_in[6];
    const float* c1w   = (const float*)d_in[7];
    const float* c1b   = (const float*)d_in[8];
    const float* c2w   = (const float*)d_in[9];
    const float* c2b   = (const float*)d_in[10];
    const float* fw_ih = (const float*)d_in[11];
    const float* fw_hh = (const float*)d_in[12];
    const float* fb    = (const float*)d_in[13];
    const float* bw_ih = (const float*)d_in[14];
    const float* bw_hh = (const float*)d_in[15];
    const float* bbias = (const float*)d_in[16];
    const float* pw1   = (const float*)d_in[17];
    const float* pb1   = (const float*)d_in[18];
    const float* pw2   = (const float*)d_in[19];
    const float* pb2   = (const float*)d_in[20];
    const float *dw1[3], *db1[3], *dw2[3], *db2[3], *dw3[3], *db3[3];
    for (int t = 0; t < 3; ++t){
        int base = 21 + t*6;
        dw1[t] = (const float*)d_in[base+0];
        db1[t] = (const float*)d_in[base+1];
        dw2[t] = (const float*)d_in[base+2];
        db2[t] = (const float*)d_in[base+3];
        dw3[t] = (const float*)d_in[base+4];
        db3[t] = (const float*)d_in[base+5];
    }
    float* outF = (float*)d_out;

    // ---- workspace bump allocator ----
    char* wsp = (char*)d_ws;
    auto alloc = [&](size_t bytes) -> void* {
        void* p = (void*)wsp;
        wsp += (bytes + 255) & ~(size_t)255;
        return p;
    };
    half_t* encIn = (half_t*)alloc((size_t)BL_*DIN_*2);
    half_t* decIn = (half_t*)alloc((size_t)BL_*DECK_*2);
    float*  c1    = (float*) alloc((size_t)BL_*CNN_*4);
    float*  cnn   = (float*) alloc((size_t)BL_*CNN_*4);
    half_t* WcatF = (half_t*)alloc((size_t)G4H_*KCAT_*2);
    half_t* WcatB = (half_t*)alloc((size_t)G4H_*KCAT_*2);
    half_t *Wd1[3], *Wd2[3], *Wd3[3];
    for (int t = 0; t < 3; ++t){
        Wd1[t] = (half_t*)alloc((size_t)H_*DECK_*2);      // [1024,2080] zero-padded
        Wd2[t] = (half_t*)alloc((size_t)512*H_*2);        // [512,1024]
        Wd3[t] = (half_t*)alloc((size_t)64*512*2);        // [64,512] zero-padded rows
    }
    float*  gates = (float*) alloc((size_t)B_*G4H_*4);
    float*  cst   = (float*) alloc((size_t)B_*H_*4);
    half_t* hping = (half_t*)alloc((size_t)B_*H_*2);
    half_t* hpong = (half_t*)alloc((size_t)B_*H_*2);
    int*    cnt   = (int*)   alloc(4*sizeof(int));
    int*    lists = (int*)   alloc((size_t)3*BL_*sizeof(int));
    half_t* h1    = (half_t*)alloc((size_t)BL_*H_*2);
    half_t* h2    = (half_t*)alloc((size_t)BL_*512*2);

    const int TB = 256;
    auto fill0 = [&](void* p, int words){
        k_fill0<<<cdiv_(words, TB), TB, 0, stream>>>((uint32_t*)p, words);
    };
    auto cvt = [&](const float* s, half_t* d, int rows, int srcK, int ld, int off){
        int tot = rows*srcK;
        k_cvt<<<cdiv_(tot, TB), TB, 0, stream>>>(s, d, rows, srcK, ld, off);
    };

    // ---- weight repack to f16 ----
    cvt(fw_ih, WcatF, G4H_, DIN_, KCAT_, 0);
    cvt(fw_hh, WcatF, G4H_, H_,   KCAT_, DIN_);
    cvt(bw_ih, WcatB, G4H_, DIN_, KCAT_, 0);
    cvt(bw_hh, WcatB, G4H_, H_,   KCAT_, DIN_);
    for (int t = 0; t < 3; ++t){
        fill0(Wd1[t], (int)((size_t)H_*DECK_*2/4));
        cvt(dw1[t], Wd1[t], H_, 2048 + PR_, DECK_, 0);
        cvt(dw2[t], Wd2[t], 512, H_, H_, 0);
        fill0(Wd3[t], (int)((size_t)64*512*2/4));
        cvt(dw3[t], Wd3[t], PR_, 512, 512, 0);
    }

    // ---- front end: embed, conv, prior ----
    k_embed<<<cdiv_(BL_*DIN_, TB), TB, 0, stream>>>(ws_i, pc_i, cembF, xembF, encIn);
    k_conv1<<<cdiv_(BL_*CNN_, TB), TB, 0, stream>>>(ws_i, cembF, c1w, c1b, c1);
    k_conv2<<<cdiv_(BL_*CNN_, TB), TB, 0, stream>>>(c1, c2w, c2b, cnn);
    k_prior<<<cdiv_(BL_, 128), 128, 0, stream>>>(ws_i, bc_i, cnn, pw1, pb1, pw2, pb2, decIn);

    // ---- BiLSTM ----
    dim3 gGrid(G4H_/128, B_/64);         // 32 x 8 blocks of 128 threads
    int pwBlocks = cdiv_(B_*H_, TB);

    // forward direction (dec_in cols 0..1023)
    fill0(cst,   B_*H_);
    fill0(hping, B_*H_*2/4);
    for (int s = 0; s < L_; ++s){
        half_t* hp = (s & 1) ? hpong : hping;
        half_t* hn = (s & 1) ? hping : hpong;
        k_lstm_gates<128><<<gGrid, 128, 0, stream>>>(encIn, hp, WcatF, fb, gates, s);
        k_lstm_point<<<pwBlocks, TB, 0, stream>>>(gates, cst, hn, decIn, s, 0);
    }
    // backward direction (dec_in cols 1024..2047)
    fill0(cst,   B_*H_);
    fill0(hping, B_*H_*2/4);
    for (int s = 0; s < L_; ++s){
        int tstep = L_ - 1 - s;
        half_t* hp = (s & 1) ? hpong : hping;
        half_t* hn = (s & 1) ? hping : hpong;
        k_lstm_gates<128><<<gGrid, 128, 0, stream>>>(encIn, hp, WcatB, bbias, gates, tstep);
        k_lstm_point<<<pwBlocks, TB, 0, stream>>>(gates, cst, hn, decIn, tstep, H_);
    }

    // ---- routed decoders ----
    fill0(outF, BL_*PR_);                 // ptype==0 rows stay zero
    fill0(cnt, 4);
    k_route<<<cdiv_(BL_, TB), TB, 0, stream>>>(pc_i, cnt, lists);

    const int MY = cdiv_(BL_, 64);        // 360 M-blocks (early-exit on count)
    for (int t = 0; t < 3; ++t){
        const int* lst = lists + (size_t)t*BL_;
        const int* cp  = cnt + (t + 1);
        k_dec_gemm<128, true,  true,  false><<<dim3(H_/128,  MY), 128, 0, stream>>>(
            decIn, DECK_, Wd1[t], DECK_, db1[t], h1, H_, nullptr, H_, lst, cp);
        k_dec_gemm<128, false, true,  false><<<dim3(512/128, MY), 128, 0, stream>>>(
            h1, H_, Wd2[t], H_, db2[t], h2, 512, nullptr, 512, lst, cp);
        k_dec_gemm<64,  false, false, true ><<<dim3(1,       MY), 128, 0, stream>>>(
            h2, 512, Wd3[t], 512, db3[t], nullptr, 0, outF, PR_, lst, cp);
    }
}
